// GIN_46840913330354
// MI455X (gfx1250) — matrix operand order repaired and checked by reading, not compiled, not run on hardware
//
#include <hip/hip_runtime.h>
#include <hip/hip_bf16.h>

#define N_NODES   100000
#define N_EDGES   1600000
#define N_GRAPHS  128
#define HID       64
#define OUT_DIM   256
#define N_LAYERS  3
#define TOT       (N_NODES * HID)          // 6,400,000
#define ROW_TILES (N_NODES / 16)           // 6250

typedef __attribute__((ext_vector_type(16))) _Float16 v16h;
typedef __attribute__((ext_vector_type(8)))  _Float16 v8h;
typedef __attribute__((ext_vector_type(8)))  float    v8f;

// ---------------- workspace layout (bytes) ----------------
#define H_OFF     0u                       // f32 [100000,64]   25,600,000
#define T0_OFF    25600000u                // f32 [100000,64]
#define T1_OFF    51200000u                // f32 [100000,64]
#define A16_OFF   76800000u                // f16 [100000,64]   12,800,000
#define WP_OFF    89600000u                // 6 * 4096 halves   49,152
#define STAT_OFF  89649408u                // 128 f32 (sum|ssq) 512
#define POOL_OFF  89649920u                // 8192 f32 gsum + 128 f32 gcnt
#define TFC_OFF   89683456u                // f32 [128,256]
#define ZFC_OFF   89814528u                // f32 [128,256]

// ---------------- elementwise helpers ----------------
__global__ __launch_bounds__(256) void k_zero(float* p, int n) {
    int i = blockIdx.x * 256 + threadIdx.x;
    if (i < n) p[i] = 0.f;
}

// h = x * proj_w + proj_b   (IN_DIM == 1)
__global__ __launch_bounds__(256) void k_proj(const float* __restrict__ x,
                                              const float* __restrict__ w,
                                              const float* __restrict__ b,
                                              float* __restrict__ H) {
    int i = blockIdx.x * 256 + threadIdx.x;
    if (i >= TOT) return;
    int n = i >> 6, j = i & 63;
    H[i] = x[n] * w[j] + b[j];
}

// out = 2*h  (== (1+eps)*h + self-loop h, before edge scatter)
__global__ __launch_bounds__(256) void k_init2h(float* __restrict__ o,
                                                const float* __restrict__ h) {
    int i = blockIdx.x * 256 + threadIdx.x;
    if (i < TOT) o[i] = 2.f * h[i];
}

__global__ __launch_bounds__(256) void k_pack16(_Float16* __restrict__ d,
                                                const float* __restrict__ s) {
    int i = blockIdx.x * 256 + threadIdx.x;
    if (i < TOT) d[i] = (_Float16)s[i];
}

// ---------------- edge scatter: one wave per edge, lane covers 2 features ----
__global__ __launch_bounds__(256) void k_scatter(const int* __restrict__ ei,
                                                 const float* __restrict__ h,
                                                 float* __restrict__ outb) {
    int wave   = (blockIdx.x * 256 + threadIdx.x) >> 5;
    int lane   = threadIdx.x & 31;
    int nwaves = (gridDim.x * 256) >> 5;
    for (int e = wave; e < N_EDGES; e += nwaves) {
        int src = ei[e];
        int dst = ei[N_EDGES + e];
        float2 v = *(const float2*)(h + src * 64 + lane * 2);
        atomicAdd(&outb[dst * 64 + lane * 2 + 0], v.x);   // no-return -> STOREcnt path
        atomicAdd(&outb[dst * 64 + lane * 2 + 1], v.y);
    }
}

// ---------------- pack W[64][64] f32 -> B-fragment layout ----------------
// Wp[((kk*4+nt)*32 + lane)*16 + j] = (f16) W[ kk*32 + (j>=8?16:0) + (lane>=16?8:0) + (j&7) ][ nt*16 + (lane&15) ]
__global__ __launch_bounds__(256) void k_packW(const float* __restrict__ W,
                                               _Float16* __restrict__ Wp) {
    int t = threadIdx.x;                 // 256 threads = 2 kk * 4 nt * 32 lanes
    int kk = t >> 7, nt = (t >> 5) & 3, lane = t & 31;
    int khalf = lane >> 4, ncol = nt * 16 + (lane & 15);
    _Float16* dst = Wp + ((kk * 4 + nt) * 32 + lane) * 16;
#pragma unroll
    for (int j = 0; j < 16; ++j) {
        int k = kk * 32 + (j >> 3) * 16 + khalf * 8 + (j & 7);
        dst[j] = (_Float16)W[k * 64 + ncol];
    }
}

// ---------------- main node GEMM: Y[rows,64] = A16[rows,64] @ W + bias ------
// One wave per 16-row tile; 4 N-tiles x 2 K-steps of v_wmma_f32_16x16x32_f16.
// Fused epilogue: per-column BN partial sums (sum, sumsq) wave->LDS->global.
__global__ __launch_bounds__(256) void k_gemm64(const _Float16* __restrict__ A16,
                                                const _Float16* __restrict__ Wp,
                                                const float* __restrict__ bias,
                                                float* __restrict__ Y,
                                                float* __restrict__ gsum,
                                                float* __restrict__ gssq) {
    __shared__ float s_sum[64], s_ssq[64];
    int tid = threadIdx.x;
    if (tid < 64) { s_sum[tid] = 0.f; s_ssq[tid] = 0.f; }
    __syncthreads();

    int lane = tid & 31;
    int wave = tid >> 5;
    int tile = blockIdx.x * 8 + wave;
    if (tile < ROW_TILES) {                       // wave-uniform guard: EXEC stays full
        int khalf = lane >> 4;                    // which K half this lane carries
        int mrow  = (tile << 4) + (lane & 15);
        const _Float16* arow = A16 + mrow * 64 + khalf * 8;

        // A fragments per ISA 16-bit A 16x32 layout:
        // a[0..7]  = K = kbase + khalf*8 + j
        // a[8..15] = K = kbase + 16 + khalf*8 + j
        v16h afrag[2];
#pragma unroll
        for (int kk = 0; kk < 2; ++kk) {
            v8h lo = *(const v8h*)(arow + kk * 32);
            v8h hi = *(const v8h*)(arow + kk * 32 + 16);
            v16h a;
#pragma unroll
            for (int j = 0; j < 8; ++j) { a[j] = lo[j]; a[8 + j] = hi[j]; }
            afrag[kk] = a;
        }

        int col = lane & 15;
#pragma unroll
        for (int nt = 0; nt < 4; ++nt) {
            v8f c = {};
#pragma unroll
            for (int kk = 0; kk < 2; ++kk) {
                v16h b = *(const v16h*)(Wp + ((kk * 4 + nt) * 32 + lane) * 16);
                c = __builtin_amdgcn_wmma_f32_16x16x32_f16(
                        false, afrag[kk], false, b, (short)0, c, false, false);
            }
            int gcol = nt * 16 + col;
            float bv = bias[gcol];
            float s = 0.f, ss = 0.f;
#pragma unroll
            for (int r = 0; r < 8; ++r) {         // C/D layout: VGPR r -> row r + 8*khalf
                float y = c[r] + bv;
                int row = (tile << 4) + r + (khalf << 3);
                Y[row * 64 + gcol] = y;
                s += y; ss += y * y;
            }
            s  += __shfl_xor(s,  16, 32);         // combine the two half-wave row groups
            ss += __shfl_xor(ss, 16, 32);
            if (lane < 16) { atomicAdd(&s_sum[gcol], s); atomicAdd(&s_ssq[gcol], ss); }
        }
    }
    __syncthreads();
    if (tid < 64) { atomicAdd(&gsum[tid], s_sum[tid]); atomicAdd(&gssq[tid], s_ssq[tid]); }
}

// ---------------- BN (training stats) + ReLU, emit f32 + f16 ----------------
__global__ __launch_bounds__(256) void k_bnrelu(const float* __restrict__ Y,
                                                const float* __restrict__ gsum,
                                                const float* __restrict__ gssq,
                                                const float* __restrict__ gamma,
                                                const float* __restrict__ beta,
                                                float* __restrict__ H,
                                                _Float16* __restrict__ H16) {
    int i = blockIdx.x * 256 + threadIdx.x;
    if (i >= TOT) return;
    int col = i & 63;
    const float invN = 1.f / (float)N_NODES;
    float mu  = gsum[col] * invN;
    float var = gssq[col] * invN - mu * mu;
    float h = gamma[col] * (Y[i] - mu) * rsqrtf(var + 1e-5f) + beta[col];
    h = fmaxf(h, 0.f);
    H[i] = h;
    H16[i] = (_Float16)h;
}

// ---------------- global mean pool ----------------
__global__ __launch_bounds__(256) void k_pool(const int* __restrict__ batch,
                                              const float* __restrict__ H,
                                              float* __restrict__ gsumP,
                                              float* __restrict__ gcnt) {
    int i = blockIdx.x * 256 + threadIdx.x;
    if (i >= TOT) return;
    int n = i >> 6, col = i & 63;
    int g = batch[n];
    atomicAdd(&gsumP[g * 64 + col], H[i]);
    if (col == 0) atomicAdd(&gcnt[g], 1.f);
}

// ---------------- FC head (tiny) ----------------
__global__ __launch_bounds__(256) void k_fc1(const float* __restrict__ gsumP,
                                             const float* __restrict__ gcnt,
                                             const float* __restrict__ w,
                                             const float* __restrict__ b,
                                             float* __restrict__ T) {
    int g = blockIdx.x, o = threadIdx.x;       // 128 x 256
    float inv = 1.f / fmaxf(gcnt[g], 1.f);
    float acc = b[o];
#pragma unroll 8
    for (int k = 0; k < 64; ++k) acc += (gsumP[g * 64 + k] * inv) * w[k * 256 + o];
    T[g * 256 + o] = acc;
}

__global__ __launch_bounds__(256) void k_fcbn(const float* __restrict__ T,
                                              const float* __restrict__ gamma,
                                              const float* __restrict__ beta,
                                              float* __restrict__ Z) {
    int o = threadIdx.x;                       // 1 block x 256 threads, col-per-thread
    float s = 0.f, ss = 0.f;
    for (int g = 0; g < N_GRAPHS; ++g) { float v = T[g * 256 + o]; s += v; ss += v * v; }
    float mu = s * (1.f / N_GRAPHS);
    float var = ss * (1.f / N_GRAPHS) - mu * mu;
    float rs = rsqrtf(var + 1e-5f);
    for (int g = 0; g < N_GRAPHS; ++g) {
        float v = gamma[o] * (T[g * 256 + o] - mu) * rs + beta[o];
        Z[g * 256 + o] = fmaxf(v, 0.f);
    }
}

__global__ __launch_bounds__(256) void k_fc2(const float* __restrict__ Z,
                                             const float* __restrict__ w,
                                             const float* __restrict__ b,
                                             float* __restrict__ out) {
    int g = blockIdx.x, o = threadIdx.x;       // 128 x 256
    float acc = b[o];
    for (int k = 0; k < 256; ++k) acc += Z[g * 256 + k] * w[k * 256 + o];
    out[g * 256 + o] = acc;
}

// ---------------- orchestration ----------------
extern "C" void kernel_launch(void* const* d_in, const int* in_sizes, int n_in,
                              void* d_out, int out_size, void* d_ws, size_t ws_size,
                              hipStream_t stream) {
    const float* x      = (const float*)d_in[0];
    const int*   ei     = (const int*)  d_in[1];
    const int*   batch  = (const int*)  d_in[2];
    const float* proj_w = (const float*)d_in[3];
    const float* proj_b = (const float*)d_in[4];
    const float* gw1    = (const float*)d_in[5];
    const float* gb1    = (const float*)d_in[6];
    const float* gg1    = (const float*)d_in[7];
    const float* gbe1   = (const float*)d_in[8];
    const float* gw2    = (const float*)d_in[9];
    const float* gb2    = (const float*)d_in[10];
    const float* gg2    = (const float*)d_in[11];
    const float* gbe2   = (const float*)d_in[12];
    const float* fcw1   = (const float*)d_in[13];
    const float* fcb1   = (const float*)d_in[14];
    const float* fcg    = (const float*)d_in[15];
    const float* fcbe   = (const float*)d_in[16];
    const float* fcw2   = (const float*)d_in[17];
    const float* fcb2   = (const float*)d_in[18];

    char* ws = (char*)d_ws;
    float*     H    = (float*)    (ws + H_OFF);
    float*     T0   = (float*)    (ws + T0_OFF);
    float*     T1   = (float*)    (ws + T1_OFF);
    _Float16*  A16  = (_Float16*) (ws + A16_OFF);
    _Float16*  WP   = (_Float16*) (ws + WP_OFF);
    float*     SSUM = (float*)    (ws + STAT_OFF);        // 64 + 64 contiguous
    float*     SSQ  = SSUM + 64;
    float*     GS   = (float*)    (ws + POOL_OFF);        // 8192 gsum + 128 gcnt
    float*     GC   = GS + N_GRAPHS * 64;
    float*     TF   = (float*)    (ws + TFC_OFF);
    float*     ZF   = (float*)    (ws + ZFC_OFF);

    const int EW_GRID   = (TOT + 255) / 256;              // 25000 blocks
    const int GEMM_GRID = (ROW_TILES + 7) / 8;            // 782 blocks

    // pack all six 64x64 weights into WMMA B-fragment layout
    for (int l = 0; l < N_LAYERS; ++l) {
        k_packW<<<1, 256, 0, stream>>>(gw1 + l * 64 * 64, WP + (2 * l + 0) * 4096);
        k_packW<<<1, 256, 0, stream>>>(gw2 + l * 64 * 64, WP + (2 * l + 1) * 4096);
    }

    k_proj<<<EW_GRID, 256, 0, stream>>>(x, proj_w, proj_b, H);

    for (int l = 0; l < N_LAYERS; ++l) {
        // out = 2h + scatter_add(h[src] -> dst)
        k_init2h<<<EW_GRID, 256, 0, stream>>>(T0, H);
        k_scatter<<<4096, 256, 0, stream>>>(ei, H, T0);
        k_pack16<<<EW_GRID, 256, 0, stream>>>(A16, T0);

        // MLP layer 1
        k_zero<<<1, 128, 0, stream>>>(SSUM, 128);
        k_gemm64<<<GEMM_GRID, 256, 0, stream>>>(A16, WP + (2 * l + 0) * 4096,
                                                gb1 + l * 64, T1, SSUM, SSQ);
        k_bnrelu<<<EW_GRID, 256, 0, stream>>>(T1, SSUM, SSQ, gg1 + l * 64,
                                              gbe1 + l * 64, H, A16);
        // MLP layer 2
        k_zero<<<1, 128, 0, stream>>>(SSUM, 128);
        k_gemm64<<<GEMM_GRID, 256, 0, stream>>>(A16, WP + (2 * l + 1) * 4096,
                                                gb2 + l * 64, T0, SSUM, SSQ);
        k_bnrelu<<<EW_GRID, 256, 0, stream>>>(T0, SSUM, SSQ, gg2 + l * 64,
                                              gbe2 + l * 64, H, A16);
    }

    // mean pool + FC head
    k_zero<<<(N_GRAPHS * 64 + N_GRAPHS + 255) / 256 + 1, 256, 0, stream>>>(GS, N_GRAPHS * 64 + N_GRAPHS);
    k_pool<<<EW_GRID, 256, 0, stream>>>(batch, H, GS, GC);
    k_fc1<<<N_GRAPHS, 256, 0, stream>>>(GS, GC, fcw1, fcb1, TF);
    k_fcbn<<<1, 256, 0, stream>>>(TF, fcg, fcbe, ZF);
    k_fc2<<<N_GRAPHS, 256, 0, stream>>>(ZF, fcw2, fcb2, (float*)d_out);
}
